// TimeSformerModel_46557445489316
// MI455X (gfx1250) — compile-verified, matrix operand order checked by reading, not run on hardware
//
#include <hip/hip_runtime.h>
#include <hip/hip_bf16.h>

// ---------------------------------------------------------------------------
// TimeSformer forward for MI455X (gfx1250, wave32, WMMA bf16 -> f32)
// ---------------------------------------------------------------------------

typedef __bf16 bf16_t;
typedef __attribute__((ext_vector_type(8)))  __bf16 v8bf;
typedef __attribute__((ext_vector_type(16))) __bf16 v16bf;
typedef __attribute__((ext_vector_type(8)))  float  v8f;

#define D_     768
#define NH_    12
#define HD_    64
#define DEPTH_ 12
#define T_     16
#define NB_    2
#define PTOT_  257
#define ROWS_  8224            // NB_*T_*PTOT_
#define SCALE_ 0.125f          // 64^-0.5

// ---------------- fragment helpers -----------------------------------------
// 16-bit A/B fragment layout (CDNA5 ISA 7.12.2): lane l holds row (l&15),
// 8 contiguous K elements at kbase=(l>>4)*8 and 8 more at kbase+16.
__device__ __forceinline__ v16bf frag_ld(const bf16_t* rowptr, int lane) {
  const int kh = (lane >> 4) * 8;
  const bf16_t* p = rowptr + kh;
  v8bf lo = *(const v8bf*)(p);
  v8bf hi = *(const v8bf*)(p + 16);
  return __builtin_shufflevector(lo, hi, 0,1,2,3,4,5,6,7,8,9,10,11,12,13,14,15);
}

__device__ __forceinline__ v8f wmma_bf16(v16bf a, v16bf b, v8f c) {
  return __builtin_amdgcn_wmma_f32_16x16x32_bf16(false, a, false, b,
                                                 (short)0, c, false, false);
}

__device__ __forceinline__ int token_row(int b, int s, int temporal) {
  if (!temporal) return b * PTOT_ + s;          // b = frame index (N*T)
  int n = b / PTOT_, p = b % PTOT_;             // b = n*PTOT + p
  return (n * T_ + s) * PTOT_ + p;              // s = time index
}

// ---------------- GEMM: C[M,N] = A[M,K](bf16) * Bt[N,K](bf16) + bias --------
// Block = 256 thr = 8 waves; block tile 128x128; wave tile 32x64 (2x4 WMMA).
// Optional residual add (Cres), fp32 out (Cout), bf16 shadow out (CoutBf).
__global__ __launch_bounds__(256) void gemm_kernel(
    const bf16_t* __restrict__ A, const bf16_t* __restrict__ Bt,
    const float* __restrict__ bias, const float* __restrict__ Cres,
    float* __restrict__ Cout, bf16_t* __restrict__ CoutBf,
    int M, int N, int K)
{
  const int lane = threadIdx.x & 31;
  const int wave = threadIdx.x >> 5;
  const int l15  = lane & 15;
  const int mb   = blockIdx.y * 128 + (wave >> 1) * 32;
  const int nb   = blockIdx.x * 128 + (wave & 1) * 64;

  v8f z = {0.f,0.f,0.f,0.f,0.f,0.f,0.f,0.f};
  v8f acc[2][4];
#pragma unroll
  for (int mi = 0; mi < 2; mi++)
#pragma unroll
    for (int nj = 0; nj < 4; nj++) acc[mi][nj] = z;

  int r0 = mb + l15;       if (r0 > M-1) r0 = M-1;
  int r1 = mb + 16 + l15;  if (r1 > M-1) r1 = M-1;
  const bf16_t* arow0 = A + (size_t)r0 * K;
  const bf16_t* arow1 = A + (size_t)r1 * K;
  const bf16_t* brow[4];
#pragma unroll
  for (int nj = 0; nj < 4; nj++)
    brow[nj] = Bt + (size_t)(nb + nj*16 + l15) * K;

  for (int kk = 0; kk < K; kk += 32) {
    __builtin_prefetch(arow0 + kk + 128, 0, 1);   // global_prefetch_b8
    __builtin_prefetch(brow[0] + kk + 128, 0, 1);
    v16bf a0 = frag_ld(arow0 + kk, lane);
    v16bf a1 = frag_ld(arow1 + kk, lane);
#pragma unroll
    for (int nj = 0; nj < 4; nj++) {
      v16bf b = frag_ld(brow[nj] + kk, lane);
      acc[0][nj] = wmma_bf16(a0, b, acc[0][nj]);
      acc[1][nj] = wmma_bf16(a1, b, acc[1][nj]);
    }
  }

  // C/D layout: lane l -> col (l&15), rows 8*(l>>4)+r for VGPR r
  const int rbase = 8 * (lane >> 4);
#pragma unroll
  for (int mi = 0; mi < 2; mi++) {
#pragma unroll
    for (int nj = 0; nj < 4; nj++) {
      const int col = nb + nj*16 + l15;
      const float bv = bias ? bias[col] : 0.f;
#pragma unroll
      for (int r = 0; r < 8; r++) {
        const int row = mb + mi*16 + rbase + r;
        if (row < M) {
          const size_t idx = (size_t)row * N + col;
          float v = acc[mi][nj][r] + bv;
          if (Cres)   v += Cres[idx];
          if (Cout)   Cout[idx] = v;
          if (CoutBf) CoutBf[idx] = (bf16_t)v;
        }
      }
    }
  }
}

// ---------------- Flash attention: one wave per 16-query tile ---------------
// qkv: [rows, 3*D] bf16 (Q at h*64, K at 768+h*64, V at 1536+h*64)
// Vt : [B, NH, HD, Spad] bf16, zero padded keys (d-major => contiguous B-frags)
// O  : [rows, D] bf16
__global__ __launch_bounds__(32) void attn_kernel(
    const bf16_t* __restrict__ qkv, const bf16_t* __restrict__ Vt,
    bf16_t* __restrict__ O, int S, int Spad, int temporal)
{
  __shared__ __align__(16) bf16_t Pld[16 * 32];
  const int lane = threadIdx.x;
  const int l15  = lane & 15;
  const int qt   = blockIdx.x;
  const int bh   = blockIdx.y;
  const int b    = bh / NH_;
  const int h    = bh % NH_;

  int qs = qt*16 + l15; if (qs > S-1) qs = S-1;
  const bf16_t* qrow = qkv + (size_t)token_row(b, qs, temporal) * (3*D_) + h*HD_;
  const v16bf aq0 = frag_ld(qrow, lane);
  const v16bf aq1 = frag_ld(qrow + 32, lane);

  float mst[8], lst[8];
#pragma unroll
  for (int r = 0; r < 8; r++) { mst[r] = -1e30f; lst[r] = 0.f; }
  v8f z = {0.f,0.f,0.f,0.f,0.f,0.f,0.f,0.f};
  v8f oacc[4] = {z, z, z, z};

  const bf16_t* vbase = Vt + (size_t)(b*NH_ + h) * HD_ * Spad;

  for (int kt = 0; kt < Spad; kt += 32) {
    // ---- scores: 16 queries x 32 keys, two 16-key halves ----
    v8f sc[2] = {z, z};
#pragma unroll
    for (int half = 0; half < 2; half++) {
      int key = kt + half*16 + l15;
      int keyc = key > S-1 ? S-1 : key;
      const bf16_t* krow =
          qkv + (size_t)token_row(b, keyc, temporal) * (3*D_) + D_ + h*HD_;
      v16bf bk0 = frag_ld(krow, lane);
      v16bf bk1 = frag_ld(krow + 32, lane);
      sc[half] = wmma_bf16(aq0, bk0, sc[half]);
      sc[half] = wmma_bf16(aq1, bk1, sc[half]);
    }
    const bool v0ok = (kt + l15) < S;
    const bool v1ok = (kt + 16 + l15) < S;

    // ---- online softmax (per-row stats; rows 8*(l>>4)+r, reduce over 16 lanes)
    float corr[8];
#pragma unroll
    for (int r = 0; r < 8; r++) {
      float s0 = v0ok ? sc[0][r] * SCALE_ : -1e30f;
      float s1 = v1ok ? sc[1][r] * SCALE_ : -1e30f;
      float mx = fmaxf(s0, s1);
#pragma unroll
      for (int m = 1; m < 16; m <<= 1) mx = fmaxf(mx, __shfl_xor(mx, m, 32));
      float mn = fmaxf(mst[r], mx);
      float c  = __expf(mst[r] - mn);
      float p0 = __expf(s0 - mn);
      float p1 = __expf(s1 - mn);
      float ps = p0 + p1;
#pragma unroll
      for (int m = 1; m < 16; m <<= 1) ps += __shfl_xor(ps, m, 32);
      lst[r] = lst[r] * c + ps;
      mst[r] = mn;
      corr[r] = c;
      const int orow = (lane >> 4) * 8 + r;
      Pld[orow*32 + l15]      = (bf16_t)p0;
      Pld[orow*32 + 16 + l15] = (bf16_t)p1;
    }
#pragma unroll
    for (int nj = 0; nj < 4; nj++)
#pragma unroll
      for (int r = 0; r < 8; r++) oacc[nj][r] *= corr[r];
    __syncthreads();

    // ---- P(16x32) @ V(32x64): re-fragment P from LDS, V from d-major Vt ----
    v16bf pa = frag_ld(&Pld[l15 * 32], lane);
#pragma unroll
    for (int nj = 0; nj < 4; nj++) {
      const bf16_t* vcol = vbase + (size_t)(nj*16 + l15) * Spad + kt;
      v16bf bv = frag_ld(vcol, lane);
      oacc[nj] = wmma_bf16(pa, bv, oacc[nj]);
    }
    __syncthreads();
  }

  // ---- normalize + scatter back to token-row layout ----
#pragma unroll
  for (int r = 0; r < 8; r++) {
    const int rl = qt*16 + (lane >> 4)*8 + r;
    if (rl < S) {
      const size_t orow = (size_t)token_row(b, rl, temporal) * D_ + h*HD_;
      const float inv = 1.f / lst[r];
#pragma unroll
      for (int nj = 0; nj < 4; nj++)
        O[orow + nj*16 + l15] = (bf16_t)(oacc[nj][r] * inv);
    }
  }
}

// ---------------- Vt builder: qkv -> [B,NH,HD,Spad] bf16, zero-padded -------
__global__ void vtbuild_kernel(const bf16_t* __restrict__ qkv,
                               bf16_t* __restrict__ Vt,
                               int B, int S, int Spad, int temporal)
{
  const size_t total = (size_t)B * NH_ * HD_ * Spad;
  for (size_t i = (size_t)blockIdx.x * blockDim.x + threadIdx.x; i < total;
       i += (size_t)gridDim.x * blockDim.x) {
    const int s  = (int)(i % Spad);
    size_t t1 = i / Spad;
    const int d  = (int)(t1 % HD_);
    size_t t2 = t1 / HD_;
    const int h  = (int)(t2 % NH_);
    const int b  = (int)(t2 / NH_);
    bf16_t v = (bf16_t)0.f;
    if (s < S)
      v = qkv[(size_t)token_row(b, s, temporal) * (3*D_) + 2*D_ + h*HD_ + d];
    Vt[i] = v;
  }
}

// ---------------- weight convert: fp32 (KxN or NxK) -> bf16 N-major (N,K) ---
__global__ void convw_kernel(const float* __restrict__ w,
                             bf16_t* __restrict__ wt,
                             int K, int N, int transpose)
{
  const size_t total = (size_t)N * K;
  for (size_t i = (size_t)blockIdx.x * blockDim.x + threadIdx.x; i < total;
       i += (size_t)gridDim.x * blockDim.x) {
    const int k = (int)(i % K);
    const int n = (int)(i / K);
    const float v = transpose ? w[(size_t)k * N + n] : w[i];
    wt[i] = (bf16_t)v;
  }
}

// ---------------- patchify: x(2,16,3,256,256) -> xp(8192, 768) bf16 ---------
__global__ void patchify_kernel(const float* __restrict__ x,
                                bf16_t* __restrict__ xp)
{
  const size_t total = (size_t)8192 * 768;
  for (size_t i = (size_t)blockIdx.x * blockDim.x + threadIdx.x; i < total;
       i += (size_t)gridDim.x * blockDim.x) {
    const int k = (int)(i % 768);
    const int m = (int)(i / 768);
    const int c  = k >> 8;
    const int py = (k >> 4) & 15;
    const int px = k & 15;
    const int nt = m >> 8;
    const int gy = (m >> 4) & 15;
    const int gx = m & 15;
    const size_t src = ((size_t)nt * 3 + c) * 65536
                     + (size_t)(gy*16 + py) * 256 + (gx*16 + px);
    xp[i] = (bf16_t)x[src];
  }
}

// ---------------- assemble h = [cls|tok] + pos + time (fp32 + bf16 shadow) --
__global__ void assemble_kernel(const float* __restrict__ tok,
                                const float* __restrict__ cls,
                                const float* __restrict__ pos,
                                const float* __restrict__ tim,
                                float* __restrict__ hf,
                                bf16_t* __restrict__ hb)
{
  const size_t total = (size_t)ROWS_ * D_;
  for (size_t i = (size_t)blockIdx.x * blockDim.x + threadIdx.x; i < total;
       i += (size_t)gridDim.x * blockDim.x) {
    const int d   = (int)(i % D_);
    const int row = (int)(i / D_);
    const int p   = row % PTOT_;
    const int nt  = row / PTOT_;
    const int t   = nt % T_;
    const float base = (p == 0) ? cls[d]
                                : tok[((size_t)nt * 256 + (p - 1)) * D_ + d];
    const float v = base + pos[(size_t)p * D_ + d] + tim[(size_t)t * D_ + d];
    hf[i] = v;
    hb[i] = (bf16_t)v;
  }
}

// ---------------- final LN (cls rows only) + head ---------------------------
__global__ __launch_bounds__(256) void head_kernel(
    const float* __restrict__ h, const float* __restrict__ g,
    const float* __restrict__ bb, const float* __restrict__ hw,
    const float* __restrict__ hb0, float* __restrict__ out)
{
  __shared__ float red[256];
  const int n = blockIdx.x, tid = threadIdx.x;
  float acc = 0.f;
  for (int t = 0; t < T_; t++) {
    const float* row = h + (size_t)((n*T_ + t) * PTOT_) * D_;
    float s = 0.f;
    for (int d = tid; d < D_; d += 256) s += row[d];
    red[tid] = s; __syncthreads();
    for (int o = 128; o > 0; o >>= 1) { if (tid < o) red[tid] += red[tid+o]; __syncthreads(); }
    const float mu = red[0] / (float)D_; __syncthreads();
    float vs = 0.f;
    for (int d = tid; d < D_; d += 256) { float df = row[d]-mu; vs += df*df; }
    red[tid] = vs; __syncthreads();
    for (int o = 128; o > 0; o >>= 1) { if (tid < o) red[tid] += red[tid+o]; __syncthreads(); }
    const float rstd = rsqrtf(red[0] / (float)D_ + 1e-5f); __syncthreads();
    float dt = 0.f;
    for (int d = tid; d < D_; d += 256)
      dt += ((row[d]-mu)*rstd*g[d] + bb[d]) * hw[d];
    red[tid] = dt; __syncthreads();
    for (int o = 128; o > 0; o >>= 1) { if (tid < o) red[tid] += red[tid+o]; __syncthreads(); }
    if (tid == 0) acc += red[0];
    __syncthreads();
  }
  if (tid == 0) out[n] = acc / (float)T_ + hb0[0];
}

// ---------------------------------------------------------------------------
// Host side
// ---------------------------------------------------------------------------
extern "C" void kernel_launch(void* const* d_in, const int* in_sizes, int n_in,
                              void* d_out, int out_size, void* d_ws, size_t ws_size,
                              hipStream_t stream)
{
  (void)in_sizes; (void)n_in; (void)out_size; (void)ws_size;
  const float* x       = (const float*)d_in[0];
  const float* patch_w = (const float*)d_in[1];
  const float* patch_b = (const float*)d_in[2];
  const float* cls_tok = (const float*)d_in[3];
  const float* pos_emb = (const float*)d_in[4];
  const float* tim_emb = (const float*)d_in[5];
  const float* wqkv_s  = (const float*)d_in[6];
  const float* bqkv_s  = (const float*)d_in[7];
  const float* wqkv_t  = (const float*)d_in[8];
  const float* bqkv_t  = (const float*)d_in[9];
  const float* wproj   = (const float*)d_in[10];
  const float* bproj   = (const float*)d_in[11];
  const float* ln_g    = (const float*)d_in[12];
  const float* ln_b    = (const float*)d_in[13];
  const float* head_w  = (const float*)d_in[14];
  const float* head_b  = (const float*)d_in[15];

  // workspace layout (bytes, 256-aligned); total ~117 MB
  const size_t OFF_HF  = 0;                      // 8224*768*4  = 25,264,128
  const size_t OFF_HBF = 25264128;               // 8224*768*2  = 12,632,064
  const size_t OFF_QKV = 37896192;               // 8224*2304*2 = 37,896,192
  const size_t OFF_TOK = OFF_QKV + 12632064;     // alias: tok fp32 (25,165,824)
  const size_t OFF_ATT = 75792384;               // 12,632,064
  const size_t OFF_VT  = 88424448;               // 25,264,128 (temporal max)
  const size_t OFF_WT  = 113688576;              // 2304*768*2 = 3,538,944

  char* ws = (char*)d_ws;
  float*  h_f   = (float*) (ws + OFF_HF);
  bf16_t* h_b   = (bf16_t*)(ws + OFF_HBF);
  bf16_t* qkv_b = (bf16_t*)(ws + OFF_QKV);
  bf16_t* xp_b  = (bf16_t*)(ws + OFF_QKV);       // alias (pre-layer only)
  float*  tok_f = (float*) (ws + OFF_TOK);       // alias (pre-layer only)
  bf16_t* att_b = (bf16_t*)(ws + OFF_ATT);
  bf16_t* vt_b  = (bf16_t*)(ws + OFF_VT);
  bf16_t* wt_b  = (bf16_t*)(ws + OFF_WT);

  // ---- patch embed + token assembly ----
  patchify_kernel<<<4096, 256, 0, stream>>>(x, xp_b);
  convw_kernel<<<2048, 256, 0, stream>>>(patch_w, wt_b, 768, 768, 0);
  gemm_kernel<<<dim3(6, 64), 256, 0, stream>>>(
      xp_b, wt_b, patch_b, nullptr, tok_f, nullptr, 8192, 768, 768);
  assemble_kernel<<<4096, 256, 0, stream>>>(tok_f, cls_tok, pos_emb, tim_emb,
                                            h_f, h_b);

  // ---- transformer layers ----
  for (int i = 0; i < DEPTH_; i++) {
    // spatial attention block
    convw_kernel<<<4096, 256, 0, stream>>>(
        wqkv_s + (size_t)i*768*2304, wt_b, 768, 2304, 1);
    gemm_kernel<<<dim3(18, 65), 256, 0, stream>>>(
        h_b, wt_b, bqkv_s + (size_t)i*2304, nullptr, nullptr, qkv_b,
        ROWS_, 2304, 768);
    vtbuild_kernel<<<4096, 256, 0, stream>>>(qkv_b, vt_b, 32, 257, 288, 0);
    attn_kernel<<<dim3(17, 32*NH_), 32, 0, stream>>>(
        qkv_b, vt_b, att_b, 257, 288, 0);
    convw_kernel<<<2048, 256, 0, stream>>>(
        wproj + (size_t)i*768*768, wt_b, 768, 768, 1);
    gemm_kernel<<<dim3(6, 65), 256, 0, stream>>>(
        att_b, wt_b, bproj + (size_t)i*768, h_f, h_f, h_b, ROWS_, 768, 768);

    // temporal attention block
    convw_kernel<<<4096, 256, 0, stream>>>(
        wqkv_t + (size_t)i*768*2304, wt_b, 768, 2304, 1);
    gemm_kernel<<<dim3(18, 65), 256, 0, stream>>>(
        h_b, wt_b, bqkv_t + (size_t)i*2304, nullptr, nullptr, qkv_b,
        ROWS_, 2304, 768);
    vtbuild_kernel<<<4096, 256, 0, stream>>>(qkv_b, vt_b, NB_*PTOT_, 16, 32, 1);
    attn_kernel<<<dim3(1, NB_*PTOT_*NH_), 32, 0, stream>>>(
        qkv_b, vt_b, att_b, 16, 32, 1);
    convw_kernel<<<2048, 256, 0, stream>>>(
        wproj + (size_t)i*768*768, wt_b, 768, 768, 1);
    gemm_kernel<<<dim3(6, 65), 256, 0, stream>>>(
        att_b, wt_b, bproj + (size_t)i*768, h_f, h_f, h_b, ROWS_, 768, 768);
  }

  // ---- final LN on cls rows + head ----
  head_kernel<<<NB_, 256, 0, stream>>>(h_f, ln_g, ln_b, head_w, head_b,
                                       (float*)d_out);
}
